// ShDictRender_3504693313894
// MI455X (gfx1250) — matrix operand order repaired and checked by reading, not compiled, not run on hardware
//
#include <hip/hip_runtime.h>
#include <hip/hip_bf16.h>
#include <math.h>

// ---------------------------------------------------------------------------
// ShDictRender on MI455X (gfx1250, wave32).
// Bin the 131072 sample points by trilinear base cell (7^3 = 343 cells).
// All points in a cell share the same 8 corner matrices M_c = atoms[:,idx_c,:]
// (32x28), so each 16-point group computes
//   data(16x28) = sum_c (diag(w_c) * Q)(16x32) @ M_c(32x28)
// as 8 chained accumulating v_wmma_f32_16x16x32_f16 ops per 16-col tile
// (K = 32 = A exactly; D = 28 padded to two N=16 tiles). The 8 corner B tiles
// are loop-invariant per block -> compiler keeps them register-resident.
// ---------------------------------------------------------------------------

typedef __attribute__((ext_vector_type(16))) _Float16 v16h;
typedef __attribute__((ext_vector_type(8)))  float    v8f;

#define B_    1024
#define NI_   256
#define NP_   131072
#define A_    32
#define R_    8
#define D_    28
#define SH_   9
#define NCELL 343     // (R-1)^3
#define DPAD  32      // D padded to 32 for WMMA N-tiles

__device__ __forceinline__ void wave_sync() {
#if defined(__has_builtin)
#if __has_builtin(__builtin_amdgcn_wave_barrier)
  __builtin_amdgcn_wave_barrier();
#else
  asm volatile("" ::: "memory");
#endif
#else
  asm volatile("" ::: "memory");
#endif
}

// ---------------------------------------------------------------- fill zeros
__global__ void k_fill_zero(float* __restrict__ p, int n) {
  int i = blockIdx.x * blockDim.x + threadIdx.x;
  int stride = gridDim.x * blockDim.x;
  for (; i < n; i += stride) p[i] = 0.0f;
}

// --------------------------------------------- atoms f32 [A][512][28] -> f16
// out layout: [cell][a][DPAD] so one corner's 32x32-half block is contiguous
// and each lane gathers its B column (16 K values) with a stride of DPAD halves.
__global__ void k_atoms_to_f16(const float* __restrict__ atoms,
                               _Float16* __restrict__ out) {
  int idx = blockIdx.x * blockDim.x + threadIdx.x;       // over 512*32*32
  if (idx >= 512 * A_ * DPAD) return;
  int d    = idx & (DPAD - 1);
  int a    = (idx >> 5) & (A_ - 1);
  int cell = idx >> 10;
  float v = (d < D_) ? atoms[((size_t)a * 512 + cell) * D_ + d] : 0.0f;
  out[idx] = (_Float16)v;
}

// ------------------------------------------------------- per-ray SH basis
__global__ void k_sh_basis(const float* __restrict__ rays_d,
                           float* __restrict__ shtab) {   // [B][16]
  int r = blockIdx.x * blockDim.x + threadIdx.x;
  if (r >= B_) return;
  float x = rays_d[r * 3 + 0], y = rays_d[r * 3 + 1], z = rays_d[r * 3 + 2];
  float inv = rsqrtf(x * x + y * y + z * z);
  x *= inv; y *= inv; z *= inv;
  const float C0 = 0.28209479177387814f;
  const float C1 = 0.4886025119029199f;
  float* o = shtab + (size_t)r * 16;
  o[0] = C0;
  o[1] = -C1 * y;
  o[2] =  C1 * z;
  o[3] = -C1 * x;
  o[4] =  1.0925484305920792f * x * y;
  o[5] = -1.0925484305920792f * y * z;
  o[6] =  0.31539156525252005f * (2.0f * z * z - x * x - y * y);
  o[7] = -1.0925484305920792f * x * z;
  o[8] =  0.5462742152960396f * (x * x - y * y);
  for (int s = 9; s < 16; ++s) o[s] = 0.0f;
}

// ------------------------------------------------- classify points into cells
__device__ __forceinline__ int base_idx(float c) {
  float cl = fminf(fmaxf(c * (float)(R_ - 1), 0.0f), (float)(R_ - 1));
  int i = (int)floorf(cl);
  return i < 0 ? 0 : (i > R_ - 2 ? R_ - 2 : i);
}

__global__ void k_classify(const float* __restrict__ intrs_pts,
                           int* __restrict__ cellid,
                           int* __restrict__ counts) {
  int p = blockIdx.x * blockDim.x + threadIdx.x;
  if (p >= NP_) return;
  int ix = base_idx(intrs_pts[p * 3 + 0]);
  int iy = base_idx(intrs_pts[p * 3 + 1]);
  int iz = base_idx(intrs_pts[p * 3 + 2]);
  int cell = (ix * 7 + iy) * 7 + iz;
  cellid[p] = cell;
  atomicAdd(&counts[cell], 1);
}

// --------------------------------------------------- tiny serial scan (343)
__global__ void k_scan(const int* __restrict__ counts,
                       int* __restrict__ offsets,
                       int* __restrict__ cursor) {
  if (blockIdx.x == 0 && threadIdx.x == 0) {
    int s = 0;
    for (int c = 0; c < NCELL; ++c) {
      offsets[c] = s;
      cursor[c]  = s;
      s += counts[c];
    }
    offsets[NCELL] = s;
  }
}

__global__ void k_scatter(const int* __restrict__ cellid,
                          int* __restrict__ cursor,
                          int* __restrict__ order) {
  int p = blockIdx.x * blockDim.x + threadIdx.x;
  if (p >= NP_) return;
  int slot = atomicAdd(&cursor[cellid[p]], 1);
  order[slot] = p;
}

// ---------------------------------------------------------------------------
// WMMA group GEMM: one block per cell, 8 waves, each wave owns 16-point groups.
// A tile  : 16 points x K=32 atoms (f16), corner weight folded into rows.
// B tiles : atoms_h[cc] 32x32 halves; lane l holds col N=l%16 (b0) / 16+l%16 (b1), 16 K values strided by DPAD;
//           loop-invariant per block -> hoisted into VGPRs by the compiler.
// C tiles : f32 16x32 accumulated across all 8 corners.
// ---------------------------------------------------------------------------
__global__ __launch_bounds__(256)
void k_group_gemm(const float* __restrict__ queries,
                  const float* __restrict__ intrs_pts,
                  const int* __restrict__ scatter_idx,
                  const _Float16* __restrict__ atoms_h,   // [512][32][32]
                  const float* __restrict__ shtab,        // [B][16]
                  const int* __restrict__ offsets,        // [NCELL+1]
                  const int* __restrict__ order,          // [NP]
                  float* __restrict__ sigma_buf,          // [B*NI]
                  float* __restrict__ rgb_buf) {          // [B*NI][4]
  __shared__ float tile[8][16][DPAD];

  const int cell = blockIdx.x;
  const int wave = threadIdx.x >> 5;
  const int lane = threadIdx.x & 31;
  const int m    = lane & 15;        // row of A / col of C within tile
  const int half = lane >> 4;        // K-half for A, M-half for C

  const int cx = cell / 49, cy = (cell / 7) % 7, cz = cell % 7;
  const int start  = offsets[cell];
  const int count  = offsets[cell + 1] - start;
  const int ngroups = (count + 15) >> 4;

  for (int g = wave; g < ngroups; g += 8) {
    const int  rel   = (g << 4) + m;
    const bool valid = rel < count;
    const int  p     = order[start + (valid ? rel : 0)];

    // fractional weights (i0 == (cx,cy,cz) by construction of the binning)
    float fx = fminf(fmaxf(intrs_pts[p * 3 + 0] * 7.0f, 0.0f), 7.0f) - (float)cx;
    float fy = fminf(fmaxf(intrs_pts[p * 3 + 1] * 7.0f, 0.0f), 7.0f) - (float)cy;
    float fz = fminf(fmaxf(intrs_pts[p * 3 + 2] * 7.0f, 0.0f), 7.0f) - (float)cz;
    const float wx[2] = { 1.0f - fx, fx };
    const float wy[2] = { 1.0f - fy, fy };
    const float wz[2] = { 1.0f - fz, fz };

    // Load this lane's A-row halves in WMMA 16-bit A layout:
    // lane<16: K in {0..7,16..23}; lane>=16: K in {8..15,24..31}.
    // Invalid rows are zeroed HERE (once), not masked per corner.
    v16h qh;
#pragma unroll
    for (int i = 0; i < 16; ++i) {
      int v = i >> 1;
      int k = ((v < 4) ? (half * 8 + (v << 1))
                       : (16 + half * 8 + ((v - 4) << 1))) + (i & 1);
      qh[i] = valid ? (_Float16)queries[(size_t)p * A_ + k] : (_Float16)0.0f;
    }

    v8f c0 = {};
    v8f c1 = {};
#pragma unroll
    for (int corner = 0; corner < 8; ++corner) {
      const int dx = (corner >> 2) & 1, dy = (corner >> 1) & 1, dz = corner & 1;
      // f32 weight product, single f32->f16 convert (selects pk_mul broadcast)
      const _Float16 hw = (_Float16)(wx[dx] * wy[dy] * wz[dz]);
      v16h hws;
#pragma unroll
      for (int i = 0; i < 16; ++i) hws[i] = hw;
      const v16h aw = qh * hws;                 // 8x v_pk_mul_f16

      const int cc = ((cx + dx) * R_ + (cy + dy)) * R_ + (cz + dz);
      const _Float16* brow = atoms_h + (size_t)cc * A_ * DPAD + m;
      v16h b0, b1;                          // col m / 16+m, K in A's table order
#pragma unroll
      for (int i = 0; i < 16; ++i) {
        int k = (i < 8) ? (half * 8 + i) : (16 + half * 8 + (i - 8));
        b0[i] = brow[k * DPAD];
        b1[i] = brow[k * DPAD + 16];
      }

      c0 = __builtin_amdgcn_wmma_f32_16x16x32_f16(
          false, aw, false, b0, (short)0, c0, false, false);
      c1 = __builtin_amdgcn_wmma_f32_16x16x32_f16(
          false, aw, false, b1, (short)0, c1, false, false);
    }

    // C layout: lane holds column N=m, rows M = r + 8*half. Transpose via LDS
    // (per-wave region; LDS ops from one wave are processed in order).
#pragma unroll
    for (int r = 0; r < 8; ++r) {
      tile[wave][r + half * 8][m]      = c0[r];
      tile[wave][r + half * 8][16 + m] = c1[r];
    }
    wave_sync();

    if (half == 0 && valid) {
      const float* row = tile[wave][m];          // data[p, 0..27]
      const int si  = scatter_idx[p];
      const int ray = si / NI_;
      sigma_buf[si] = row[27];
      const float* sh = shtab + (size_t)ray * 16;
#pragma unroll
      for (int c = 0; c < 3; ++c) {
        float acc = 0.0f;
#pragma unroll
        for (int s = 0; s < SH_; ++s) acc += sh[s] * row[c * SH_ + s];
        rgb_buf[(size_t)si * 4 + c] = acc;
      }
    }
    wave_sync();
  }
}

// ---------------------------------------------------- per-ray volume render
__global__ void k_render(const float* __restrict__ rays_d,
                         const float* __restrict__ inter,      // [B][NI+1]
                         const float* __restrict__ sigma_buf,  // [B*NI]
                         const float* __restrict__ rgb_buf,    // [B*NI][4]
                         float* __restrict__ out_rgb,          // [B][3]
                         float* __restrict__ out_alpha,        // [B][NI]
                         float* __restrict__ out_depth) {      // [B]
  int r = blockIdx.x * blockDim.x + threadIdx.x;
  if (r >= B_) return;
  float dx = rays_d[r * 3 + 0], dy = rays_d[r * 3 + 1], dz = rays_d[r * 3 + 2];
  float nrm = sqrtf(dx * dx + dy * dy + dz * dz);
  float trans = 1.0f, acc = 0.0f, depth = 0.0f;
  float a0 = 0.0f, a1 = 0.0f, a2 = 0.0f;
  float t0 = inter[(size_t)r * (NI_ + 1)];
  for (int i = 0; i < NI_; ++i) {
    float t1 = inter[(size_t)r * (NI_ + 1) + i + 1];
    float dist  = (t1 - t0) * nrm;
    float sg    = sigma_buf[(size_t)r * NI_ + i];
    sg = sg > 0.0f ? sg : 0.0f;
    float alpha = 1.0f - __expf(-sg * dist);
    out_alpha[(size_t)r * NI_ + i] = alpha;
    float w = alpha * trans;
    trans *= (1.0f - alpha + 1e-10f);
    const float* rb = rgb_buf + ((size_t)r * NI_ + i) * 4;
    a0 += w * (1.0f / (1.0f + __expf(-rb[0])));
    a1 += w * (1.0f / (1.0f + __expf(-rb[1])));
    a2 += w * (1.0f / (1.0f + __expf(-rb[2])));
    acc   += w;
    depth += w * 0.5f * (t0 + t1);
    t0 = t1;
  }
  float bg = 1.0f - acc;   // WHITE_BKGD
  out_rgb[(size_t)r * 3 + 0] = a0 + bg;
  out_rgb[(size_t)r * 3 + 1] = a1 + bg;
  out_rgb[(size_t)r * 3 + 2] = a2 + bg;
  out_depth[r] = depth;
}

// ---------------------------------------------------------------------------
extern "C" void kernel_launch(void* const* d_in, const int* in_sizes, int n_in,
                              void* d_out, int out_size, void* d_ws, size_t ws_size,
                              hipStream_t stream) {
  const float* rays_d      = (const float*)d_in[0];  // [B,3]
  const float* queries     = (const float*)d_in[1];  // [NP,A]
  const float* intrs_pts   = (const float*)d_in[2];  // [NP,3]
  const float* inter       = (const float*)d_in[3];  // [B,NI+1]
  const float* atoms       = (const float*)d_in[4];  // [A,512,D]
  const int*   scatter_idx = (const int*)d_in[5];    // [NP]

  // Workspace layout (256B-aligned slices)
  char* ws = (char*)d_ws;
  size_t off = 0;
  auto alloc = [&](size_t bytes) {
    char* p = ws + off;
    off += (bytes + 255) & ~(size_t)255;
    return p;
  };
  _Float16* atoms_h  = (_Float16*)alloc((size_t)512 * A_ * DPAD * 2);   // 1 MB
  float*    shtab    = (float*)   alloc((size_t)B_ * 16 * 4);           // 64 KB
  int*      counts   = (int*)     alloc((size_t)(NCELL + 1) * 4);
  int*      offsets  = (int*)     alloc((size_t)(NCELL + 1) * 4);
  int*      cursor   = (int*)     alloc((size_t)(NCELL + 1) * 4);
  int*      cellid   = (int*)     alloc((size_t)NP_ * 4);               // 512 KB
  int*      order    = (int*)     alloc((size_t)NP_ * 4);               // 512 KB
  float*    sigma_b  = (float*)   alloc((size_t)B_ * NI_ * 4);          // 1 MB
  float*    rgb_b    = (float*)   alloc((size_t)B_ * NI_ * 4 * 4);      // 4 MB
  (void)ws_size; (void)n_in; (void)in_sizes; (void)out_size;

  float* out_rgb   = (float*)d_out;               // B*3
  float* out_alpha = out_rgb + (size_t)B_ * 3;    // B*NI
  float* out_depth = out_alpha + (size_t)B_ * NI_;

  // Zero scratch used via scatter/atomics (ws is poisoned, not re-poisoned).
  k_fill_zero<<<4, 256, 0, stream>>>((float*)counts, NCELL + 1);
  k_fill_zero<<<512, 256, 0, stream>>>(sigma_b, B_ * NI_);
  k_fill_zero<<<2048, 256, 0, stream>>>(rgb_b, B_ * NI_ * 4);

  // Preprocess
  k_atoms_to_f16<<<(512 * A_ * DPAD + 255) / 256, 256, 0, stream>>>(atoms, atoms_h);
  k_sh_basis<<<(B_ + 255) / 256, 256, 0, stream>>>(rays_d, shtab);

  // Bin points by trilinear base cell
  k_classify<<<NP_ / 256, 256, 0, stream>>>(intrs_pts, cellid, counts);
  k_scan<<<1, 1, 0, stream>>>(counts, offsets, cursor);
  k_scatter<<<NP_ / 256, 256, 0, stream>>>(cellid, cursor, order);

  // WMMA einsum: one block per cell, 8 waves each
  k_group_gemm<<<NCELL, 256, 0, stream>>>(queries, intrs_pts, scatter_idx,
                                          atoms_h, shtab, offsets, order,
                                          sigma_b, rgb_b);

  // Per-ray volume rendering
  k_render<<<(B_ + 255) / 256, 256, 0, stream>>>(rays_d, inter, sigma_b, rgb_b,
                                                 out_rgb, out_alpha, out_depth);
}
